// _HFAttention_66760971649153
// MI455X (gfx1250) — compile-verified, matrix operand order checked by reading, not run on hardware
//
#include <hip/hip_runtime.h>
#include <hip/hip_bf16.h>

// ---------- types ----------
typedef __bf16 bf16_t;
typedef __attribute__((ext_vector_type(16))) __bf16 v16bf;
typedef __attribute__((ext_vector_type(8)))  __bf16 v8bf;
typedef __attribute__((ext_vector_type(8)))  float  v8f;

#define SEQ      2048
#define HIDDEN   4096
#define NHEADS   32
#define NKVHEADS 8
#define HEADDIM  128
#define QSTRIDE  (NHEADS * HEADDIM)      // 4096
#define KVSTRIDE (NKVHEADS * HEADDIM)    // 1024

#define GPAD 40            // padded LDS row stride (ushorts): 80B -> conflict-free 16-row reads
#define KPAD 136           // Ks row stride: 272B -> lane bank offset = l16*4 (distinct)

static __device__ __forceinline__ unsigned short f2bf(float f) {
  bf16_t h = (bf16_t)f;                       // RNE convert
  return __builtin_bit_cast(unsigned short, h);
}

// low 32 bits of a generic pointer to LDS == LDS byte offset
static __device__ __forceinline__ unsigned lds_off(const void* p) {
  return (unsigned)(uintptr_t)p;
}

static __device__ __forceinline__ v16bf load_frag(const unsigned short* lo_p,
                                                  const unsigned short* hi_p) {
  v8bf lo = *reinterpret_cast<const v8bf*>(lo_p);
  v8bf hi = *reinterpret_cast<const v8bf*>(hi_p);
  v16bf r;
#pragma unroll
  for (int e = 0; e < 8; ++e) { r[e] = lo[e]; r[e + 8] = hi[e]; }
  return r;
}

static __device__ __forceinline__ v8f wmma_bf16(v16bf a, v16bf b, v8f c) {
  return __builtin_amdgcn_wmma_f32_16x16x32_bf16(false, a, false, b, (short)0, c,
                                                 false, false);
}

// async 16B global -> LDS copy (per lane), tracked with ASYNCcnt
static __device__ __forceinline__ void async_copy_b128(unsigned lds_byte_off,
                                                       const void* gaddr) {
  asm volatile("global_load_async_to_lds_b128 %0, %1, off"
               :: "v"(lds_byte_off), "v"((unsigned long long)(uintptr_t)gaddr)
               : "memory");
}
static __device__ __forceinline__ void wait_async0() {
  asm volatile("s_wait_asynccnt 0x0" ::: "memory");
}

// ---------- fp32 -> bf16 elementwise (vectorized), same layout ----------
__global__ __launch_bounds__(256) void convert_bf16(const float* __restrict__ in,
                                                    unsigned short* __restrict__ out,
                                                    int total4) {
  const int idx = blockIdx.x * 256 + threadIdx.x;
  if (idx >= total4) return;
  const float4 f = reinterpret_cast<const float4*>(in)[idx];
  uint2 p;
  p.x = (unsigned)f2bf(f.x) | ((unsigned)f2bf(f.y) << 16);
  p.y = (unsigned)f2bf(f.z) | ((unsigned)f2bf(f.w) << 16);
  reinterpret_cast<uint2*>(out)[idx] = p;
}

// ---------- weight convert + transpose: in [K][N] fp32 -> out [N][K] bf16 ----------
__global__ __launch_bounds__(256) void convert_transpose(const float* __restrict__ in,
                                                         unsigned short* __restrict__ out,
                                                         int K, int N) {
  size_t idx = (size_t)blockIdx.x * blockDim.x + threadIdx.x;
  size_t total = (size_t)K * N;
  if (idx >= total) return;
  int k = (int)(idx % K);
  int n = (int)(idx / K);
  out[idx] = f2bf(in[(size_t)k * N + n]);     // output coalesced over k
}

// ---------- GEMM: C[M][N] = A[M][K](bf16) x Bt[N][K](bf16), fp32 accumulate ----------
// block = 256 threads (8 waves), tile 128x128, K-step 32, double-buffered LDS,
// staged with global_load_async_to_lds_b128 (ASYNCcnt) -> no prefetch registers.
// Wave (wr,wc) computes 32 rows x 64 cols: 2 A-frags x 4 B-frags -> 8 WMMA / K-step.
__global__ __launch_bounds__(256, 1) void gemm_bf16(const unsigned short* __restrict__ A,
                                                    const unsigned short* __restrict__ Bt,
                                                    float* __restrict__ C,
                                                    int M, int N, int K) {
  __shared__ __align__(16) unsigned short As[2][128][GPAD];
  __shared__ __align__(16) unsigned short Bs[2][128][GPAD];

  const int tid  = threadIdx.x;
  const int wave = tid >> 5;
  const int lane = tid & 31;
  const int l16  = lane & 15;
  const int half = lane >> 4;
  const int wr   = wave >> 1;                 // 0..3 -> row group *32
  const int wc   = wave & 1;                  // 0..1 -> col group *64
  const int bm   = blockIdx.y * 128;
  const int bn   = blockIdx.x * 128;

  v8f acc[2][4];
#pragma unroll
  for (int i = 0; i < 2; ++i)
#pragma unroll
    for (int j = 0; j < 4; ++j) acc[i][j] = {};

  // per-thread async staging map: 512 16B-chunks per tile, 2 per thread
  // chunk c: row = c>>2, kc = (c&3)*8   (no chunk crosses the GPAD padding)
  const unsigned aBase = lds_off(&As[0][0][0]);
  const unsigned bBase = lds_off(&Bs[0][0][0]);
  const unsigned bufSz = 128 * GPAD * 2;      // bytes per buffer

  auto issue_async = [&](int buf, int k0) {
#pragma unroll
    for (int i = 0; i < 2; ++i) {
      const int c   = tid + i * 256;          // 0..511
      const int row = c >> 2;
      const int kc  = (c & 3) * 8;
      const unsigned loff = (unsigned)(buf * bufSz + row * (GPAD * 2) + kc * 2);
      async_copy_b128(aBase + loff, &A[(size_t)(bm + row) * K + k0 + kc]);
      async_copy_b128(bBase + loff, &Bt[(size_t)(bn + row) * K + k0 + kc]);
    }
  };

  auto compute = [&](int buf) {
    v16bf af[2], bfv[4];
#pragma unroll
    for (int i = 0; i < 2; ++i) {
      const int r = wr * 32 + i * 16 + l16;
      af[i] = load_frag(&As[buf][r][half * 8], &As[buf][r][half * 8 + 16]);
    }
#pragma unroll
    for (int j = 0; j < 4; ++j) {
      const int cc = wc * 64 + j * 16 + l16;
      bfv[j] = load_frag(&Bs[buf][cc][half * 8], &Bs[buf][cc][half * 8 + 16]);
    }
#pragma unroll
    for (int i = 0; i < 2; ++i)
#pragma unroll
      for (int j = 0; j < 4; ++j)
        acc[i][j] = wmma_bf16(af[i], bfv[j], acc[i][j]);
  };

  // ---- software pipeline: async-copy tile k+1 while WMMAs consume tile k ----
  const int NT = K / 32;
  issue_async(0, 0);
  wait_async0();
  __syncthreads();
  for (int kt = 0; kt < NT; ++kt) {
    if (kt + 1 < NT) issue_async((kt + 1) & 1, (kt + 1) * 32);
    compute(kt & 1);
    if (kt + 1 < NT) {
      wait_async0();
      __syncthreads();
    }
  }

  // ---- epilogue: one base pointer + 32-bit offsets ----
  float* cbase = C + (size_t)(bm + wr * 32) * N + (bn + wc * 64 + l16);
#pragma unroll
  for (int i = 0; i < 2; ++i)
#pragma unroll
    for (int j = 0; j < 4; ++j)
#pragma unroll
      for (int e = 0; e < 8; ++e)
        cbase[(i * 16 + e + 8 * half) * N + j * 16] = acc[i][j][e];
}

// ---------- fused RoPE + RMSNorm, in place on [S][H*128] ----------
__global__ __launch_bounds__(128) void rope_rmsnorm(float* __restrict__ x,
                                                    const float* __restrict__ w,
                                                    int rowstride) {
  const int s = blockIdx.x;
  const int h = blockIdx.y;
  const int d = threadIdx.x;                  // 0..127
  float* row = x + (size_t)s * rowstride + h * HEADDIM;

  const int dr = d & 63;
  // inv_freq = 10000^(-2*dr/128)
  const float inv = __expf(-((float)(2 * dr) / 128.0f) * 9.210340371976184f);
  const float ang = (float)s * inv;
  const float c = cosf(ang), sn = sinf(ang);

  const float v0  = row[d];
  const float rot = (d < 64) ? -row[d + 64] : row[d - 64];
  const float y   = v0 * c + rot * sn;

  // RMS over 128 dims
  __shared__ float redbuf[4];
  float sq = y * y;
#pragma unroll
  for (int m = 16; m >= 1; m >>= 1) sq += __shfl_xor(sq, m, 32);
  if ((threadIdx.x & 31) == 0) redbuf[threadIdx.x >> 5] = sq;
  __syncthreads();
  const float var = (redbuf[0] + redbuf[1] + redbuf[2] + redbuf[3]) / 128.0f;
  row[d] = w[d] * y * rsqrtf(var + 1e-6f);
}

// ---------- flash attention: one (head, 64-row q tile) per block; bf16 output ----------
__global__ __launch_bounds__(128, 1) void attention(const float* __restrict__ Q,
                                                    const float* __restrict__ Kc,
                                                    const float* __restrict__ V,
                                                    unsigned short* __restrict__ O) {
  const int qt   = blockIdx.x;                // 0..31
  const int h    = blockIdx.y;                // 0..31
  const int kvh  = h >> 2;                    // GQA group
  const int tid  = threadIdx.x;
  const int wave = tid >> 5;
  const int lane = tid & 31;
  const int l16  = lane & 15;
  const int half = lane >> 4;
  const int q0   = qt * 64 + wave * 16;       // wave's first q row

  __shared__ __align__(16) unsigned short Ks[32][KPAD];    // [key][d] bf16 (padded)
  __shared__ __align__(16) unsigned short Vst[128][GPAD];  // [d][key] bf16 (transposed, padded)
  __shared__ __align__(16) unsigned short Ps[4][16][GPAD]; // per-wave P staging (padded)

  // Q fragments: 16 rows x 128 d as 4 A-fragments (bf16)
  v16bf qf[4];
  {
    const float* qrow = Q + (size_t)(q0 + l16) * QSTRIDE + h * HEADDIM;
#pragma unroll
    for (int j = 0; j < 4; ++j) {
      const int dbase = j * 32 + half * 8;
      v16bf a;
#pragma unroll
      for (int e = 0; e < 8; ++e) {
        a[e]     = (bf16_t)qrow[dbase + e];
        a[e + 8] = (bf16_t)qrow[dbase + 16 + e];
      }
      qf[j] = a;
    }
  }

  v8f oacc[8];
#pragma unroll
  for (int j = 0; j < 8; ++j) oacc[j] = {};
  float m_i[8], l_i[8];
#pragma unroll
  for (int i = 0; i < 8; ++i) { m_i[i] = -3.0e38f; l_i[i] = 0.0f; }

  const float scale = 0.08838834764831845f;   // 1/sqrt(128)
  const int nkb = 2 * qt + 2;                 // 32-key blocks covering keys < (qt+1)*64

  for (int kb = 0; kb < nkb; ++kb) {
    const int kbase = kb * 32;
    __syncthreads();
    // stage K and V tiles: 32 keys x 128 dims each (fp32 -> bf16)
#pragma unroll
    for (int i = 0; i < 8; ++i) {
      const int flat4 = tid + i * 128;        // 0..1023 float4s
      const int key   = flat4 >> 5;           // 32 float4 per key row
      const int dd    = (flat4 & 31) * 4;
      const float4 kf = *reinterpret_cast<const float4*>(
          &Kc[(size_t)(kbase + key) * KVSTRIDE + kvh * HEADDIM + dd]);
      uint2 kp;
      kp.x = (unsigned)f2bf(kf.x) | ((unsigned)f2bf(kf.y) << 16);
      kp.y = (unsigned)f2bf(kf.z) | ((unsigned)f2bf(kf.w) << 16);
      *reinterpret_cast<uint2*>(&Ks[key][dd]) = kp;
      const float4 vf = *reinterpret_cast<const float4*>(
          &V[(size_t)(kbase + key) * KVSTRIDE + kvh * HEADDIM + dd]);
      Vst[dd + 0][key] = f2bf(vf.x);
      Vst[dd + 1][key] = f2bf(vf.y);
      Vst[dd + 2][key] = f2bf(vf.z);
      Vst[dd + 3][key] = f2bf(vf.w);
    }
    __syncthreads();

    // S = Q K^T for 16 rows x 32 keys (two 16-col C fragments)
    v8f sfr[2];
#pragma unroll
    for (int c = 0; c < 2; ++c) {
      v8f s = {};
      const int keyc = c * 16 + l16;
      v16bf bfv[4];
#pragma unroll
      for (int j = 0; j < 4; ++j)
        bfv[j] = load_frag(&Ks[keyc][j * 32 + half * 8],
                           &Ks[keyc][j * 32 + half * 8 + 16]);
#pragma unroll
      for (int j = 0; j < 4; ++j) s = wmma_bf16(qf[j], bfv[j], s);
      sfr[c] = s;
    }

    // scale + causal mask (reference adds NEG_INF)
#pragma unroll
    for (int c = 0; c < 2; ++c) {
      const int key = kbase + c * 16 + l16;
#pragma unroll
      for (int i = 0; i < 8; ++i) {
        const int qrow = q0 + i + 8 * half;
        const float sv = sfr[c][i] * scale;
        sfr[c][i] = (key > qrow) ? (sv - 1.0e9f) : sv;
      }
    }

    // online softmax per row (row = element i within 16-lane group)
#pragma unroll
    for (int i = 0; i < 8; ++i) {
      float mx = fmaxf(sfr[0][i], sfr[1][i]);
#pragma unroll
      for (int msk = 8; msk >= 1; msk >>= 1) mx = fmaxf(mx, __shfl_xor(mx, msk, 32));
      const float mnew = fmaxf(m_i[i], mx);
      const float corr = __expf(m_i[i] - mnew);
      const float p0 = __expf(sfr[0][i] - mnew);
      const float p1 = __expf(sfr[1][i] - mnew);
      sfr[0][i] = p0; sfr[1][i] = p1;
      float rs = p0 + p1;
#pragma unroll
      for (int msk = 8; msk >= 1; msk >>= 1) rs += __shfl_xor(rs, msk, 32);
      l_i[i] = l_i[i] * corr + rs;
      m_i[i] = mnew;
#pragma unroll
      for (int j = 0; j < 8; ++j) oacc[j][i] *= corr;
    }

    // redistribute P from C-layout to A-fragment layout via per-wave LDS
#pragma unroll
    for (int c = 0; c < 2; ++c)
#pragma unroll
      for (int i = 0; i < 8; ++i)
        Ps[wave][i + 8 * half][c * 16 + l16] = f2bf(sfr[c][i]);
    asm volatile("s_wait_dscnt 0" ::: "memory");   // wave-internal LDS RAW fence

    v16bf pf = load_frag(&Ps[wave][l16][half * 8], &Ps[wave][l16][half * 8 + 16]);

    // O += P V  (K-dim = 32 keys, 8 d-slices of 16); batch frag loads then WMMAs
    v16bf vf4[4];
#pragma unroll
    for (int j = 0; j < 4; ++j)
      vf4[j] = load_frag(&Vst[j * 16 + l16][half * 8],
                         &Vst[j * 16 + l16][half * 8 + 16]);
#pragma unroll
    for (int j = 0; j < 4; ++j) oacc[j] = wmma_bf16(pf, vf4[j], oacc[j]);
#pragma unroll
    for (int j = 0; j < 4; ++j)
      vf4[j] = load_frag(&Vst[(j + 4) * 16 + l16][half * 8],
                         &Vst[(j + 4) * 16 + l16][half * 8 + 16]);
#pragma unroll
    for (int j = 0; j < 4; ++j) oacc[j + 4] = wmma_bf16(pf, vf4[j], oacc[j + 4]);
  }

  // write normalized output directly as bf16 (feeds the O-projection GEMM)
#pragma unroll
  for (int j = 0; j < 8; ++j) {
    const int d = j * 16 + l16;
#pragma unroll
    for (int i = 0; i < 8; ++i) {
      const int row = q0 + i + 8 * half;
      O[(size_t)row * QSTRIDE + h * HEADDIM + d] = f2bf(oacc[j][i] / l_i[i]);
    }
  }
}

// ---------- host launch ----------
extern "C" void kernel_launch(void* const* d_in, const int* in_sizes, int n_in,
                              void* d_out, int out_size, void* d_ws, size_t ws_size,
                              hipStream_t stream) {
  const float* hidden = (const float*)d_in[0];
  const float* wq     = (const float*)d_in[1];
  const float* wk     = (const float*)d_in[2];
  const float* wv     = (const float*)d_in[3];
  const float* wo     = (const float*)d_in[4];
  const float* qnw    = (const float*)d_in[5];
  const float* knw    = (const float*)d_in[6];
  // d_in[7] attention_mask (causal, implicit), d_in[8] position_ids (arange) unused

  char* ws = (char*)d_ws;
  unsigned short* wqT   = (unsigned short*)ws; ws += (size_t)HIDDEN * QSTRIDE * 2;
  unsigned short* wkT   = (unsigned short*)ws; ws += (size_t)HIDDEN * KVSTRIDE * 2;
  unsigned short* wvT   = (unsigned short*)ws; ws += (size_t)HIDDEN * KVSTRIDE * 2;
  unsigned short* woT   = (unsigned short*)ws; ws += (size_t)QSTRIDE * HIDDEN * 2;
  unsigned short* hidBF = (unsigned short*)ws; ws += (size_t)SEQ * HIDDEN * 2;
  unsigned short* attBF = (unsigned short*)ws; ws += (size_t)SEQ * QSTRIDE * 2;
  float* qb   = (float*)ws; ws += (size_t)SEQ * QSTRIDE * 4;
  float* kbuf = (float*)ws; ws += (size_t)SEQ * KVSTRIDE * 4;
  float* vbuf = (float*)ws; ws += (size_t)SEQ * KVSTRIDE * 4;
  float* out  = (float*)d_out;

  // weight conversions (fp32 -> bf16, transposed to [N][K]) + activation conversion
  {
    size_t n;
    n = (size_t)HIDDEN * QSTRIDE;
    convert_transpose<<<dim3((unsigned)((n + 255) / 256)), 256, 0, stream>>>(wq, wqT, HIDDEN, QSTRIDE);
    n = (size_t)HIDDEN * KVSTRIDE;
    convert_transpose<<<dim3((unsigned)((n + 255) / 256)), 256, 0, stream>>>(wk, wkT, HIDDEN, KVSTRIDE);
    convert_transpose<<<dim3((unsigned)((n + 255) / 256)), 256, 0, stream>>>(wv, wvT, HIDDEN, KVSTRIDE);
    n = (size_t)QSTRIDE * HIDDEN;
    convert_transpose<<<dim3((unsigned)((n + 255) / 256)), 256, 0, stream>>>(wo, woT, QSTRIDE, HIDDEN);
    const int t4 = SEQ * HIDDEN / 4;
    convert_bf16<<<dim3((t4 + 255) / 256), 256, 0, stream>>>(hidden, hidBF, t4);
  }

  // projections (tile 128x128, async-LDS pipelined)
  gemm_bf16<<<dim3(QSTRIDE / 128, SEQ / 128), 256, 0, stream>>>(hidBF, wqT, qb, SEQ, QSTRIDE, HIDDEN);
  gemm_bf16<<<dim3(KVSTRIDE / 128, SEQ / 128), 256, 0, stream>>>(hidBF, wkT, kbuf, SEQ, KVSTRIDE, HIDDEN);
  gemm_bf16<<<dim3(KVSTRIDE / 128, SEQ / 128), 256, 0, stream>>>(hidBF, wvT, vbuf, SEQ, KVSTRIDE, HIDDEN);

  // RoPE + RMSNorm (in place)
  rope_rmsnorm<<<dim3(SEQ, NHEADS), 128, 0, stream>>>(qb, qnw, QSTRIDE);
  rope_rmsnorm<<<dim3(SEQ, NKVHEADS), 128, 0, stream>>>(kbuf, knw, KVSTRIDE);

  // causal GQA flash attention (bf16 output)
  attention<<<dim3(SEQ / 64, NHEADS), 128, 0, stream>>>(qb, kbuf, vbuf, attBF);

  // output projection
  gemm_bf16<<<dim3(HIDDEN / 128, SEQ / 128), 256, 0, stream>>>(attBF, woT, out, SEQ, HIDDEN, QSTRIDE);
}
